// CausalSelfAttention_57827439673490
// MI455X (gfx1250) — matrix operand order repaired and checked by reading, not compiled, not run on hardware
//
#include <hip/hip_runtime.h>

// MI455X / gfx1250, wave32, WMMA f16 16x16x32 with f32 accumulate.
// Causal self-attention: qkv GEMM -> flash attention -> out projection.
// Round 2: K/V tile staging now uses GLOBAL_LOAD_ASYNC_TO_LDS_B128
// (ASYNCcnt-tracked async copy engine) instead of VGPR round-trips.

typedef _Float16 v16h __attribute__((ext_vector_type(16)));
typedef float    v8f  __attribute__((ext_vector_type(8)));

constexpr int kBatch = 2;
constexpr int kT     = 2048;
constexpr int kC     = 768;
constexpr int kH     = 12;
constexpr int kD     = 64;
constexpr int kNqkv  = 3 * kC;      // 2304
constexpr int kM     = kBatch * kT; // 4096

static __device__ __forceinline__ v8f wmma_f16(v16h a, v16h b, v8f c) {
  // (neg_a, A, neg_b, B, c_mod, C, reuse_a, reuse_b)
  return __builtin_amdgcn_wmma_f32_16x16x32_f16(false, a, false, b, (short)0, c,
                                                false, false);
}

// A-matrix (16x32 f16) K index for VGPR-half j (0..7), lane half (0/1).
// Lanes 0-15: K = {0..7, 16..23}; lanes 16-31: K = {8..15, 24..31}.
static __device__ __forceinline__ int a_kidx(int j, int half) {
  return ((j < 4) ? (2 * j) : (2 * j + 8)) + 8 * half;
}
// B-matrix (32x16 f16) K index: same as A; lanes 0-15 K = {0..7, 16..23}, lanes 16-31 K = {8..15, 24..31}.
static __device__ __forceinline__ int b_kidx(int j, int half) {
  return ((j < 4) ? (2 * j) : (2 * j + 8)) + 8 * half;
}

// Async 16-byte global -> LDS copy (per-lane addresses, ASYNCcnt-tracked).
// lds_off is the wave-relative LDS byte offset (low 32 bits of generic ptr).
static __device__ __forceinline__ void async_copy_b128(uint32_t lds_off,
                                                       const void* gptr) {
  asm volatile("global_load_async_to_lds_b128 %0, %1, off"
               :
               : "v"(lds_off), "v"((unsigned long long)(uintptr_t)gptr)
               : "memory");
}

// ---------------------------------------------------------------------------
// Kernel 1: qkv = x @ qkv_w^T + qkv_b, scattered to f16 Q/K/V in [B,H,T,D].
// Block: 256 thr (8 waves), tile 128(M) x 64(N), K-step 32.
// ---------------------------------------------------------------------------
__global__ __launch_bounds__(256) void qkv_gemm_kernel(
    const float* __restrict__ x, const float* __restrict__ qkv_w,
    const float* __restrict__ qkv_b, _Float16* __restrict__ qbuf,
    _Float16* __restrict__ kbuf, _Float16* __restrict__ vbuf) {
  __shared__ __align__(16) _Float16 Xs[128][40];
  __shared__ __align__(16) _Float16 Ws[64][40];

  const int tid  = threadIdx.x;
  const int wave = tid >> 5;
  const int lane = tid & 31;
  const int half = lane >> 4;
  const int l16  = lane & 15;
  const int m0   = blockIdx.x * 128;
  const int n0   = blockIdx.y * 64;
  const int waveM = wave >> 1;  // 0..3
  const int waveN = wave & 1;   // 0..1

  v8f acc[2][2] = {};

  const int ldr  = tid >> 3;       // 0..31
  const int ldc4 = (tid & 7) * 4;  // 0,4,...,28

  for (int kk0 = 0; kk0 < kC; kk0 += 32) {
#pragma unroll
    for (int p = 0; p < 4; ++p) {
      const int row = p * 32 + ldr;
      const float4 v = *reinterpret_cast<const float4*>(
          &x[(size_t)(m0 + row) * kC + kk0 + ldc4]);
      Xs[row][ldc4 + 0] = (_Float16)v.x;
      Xs[row][ldc4 + 1] = (_Float16)v.y;
      Xs[row][ldc4 + 2] = (_Float16)v.z;
      Xs[row][ldc4 + 3] = (_Float16)v.w;
    }
#pragma unroll
    for (int p = 0; p < 2; ++p) {
      const int row = p * 32 + ldr;
      const float4 v = *reinterpret_cast<const float4*>(
          &qkv_w[(size_t)(n0 + row) * kC + kk0 + ldc4]);
      Ws[row][ldc4 + 0] = (_Float16)v.x;
      Ws[row][ldc4 + 1] = (_Float16)v.y;
      Ws[row][ldc4 + 2] = (_Float16)v.z;
      Ws[row][ldc4 + 3] = (_Float16)v.w;
    }
    __syncthreads();

    v16h afr[2], bfr[2];
#pragma unroll
    for (int i = 0; i < 2; ++i) {
      const int row = waveM * 32 + i * 16 + l16;
#pragma unroll
      for (int j = 0; j < 8; ++j) {
        const int k = a_kidx(j, half);
        afr[i][2 * j]     = Xs[row][k];
        afr[i][2 * j + 1] = Xs[row][k + 1];
      }
    }
#pragma unroll
    for (int i = 0; i < 2; ++i) {
      const int col = waveN * 32 + i * 16 + l16;
#pragma unroll
      for (int j = 0; j < 8; ++j) {
        const int k = b_kidx(j, half);
        bfr[i][2 * j]     = Ws[col][k];
        bfr[i][2 * j + 1] = Ws[col][k + 1];
      }
    }
#pragma unroll
    for (int i = 0; i < 2; ++i)
#pragma unroll
      for (int jn = 0; jn < 2; ++jn)
        acc[i][jn] = wmma_f16(afr[i], bfr[jn], acc[i][jn]);
    __syncthreads();
  }

  // Epilogue: bias add + scatter to [B,H,T,D] f16 buffers.
#pragma unroll
  for (int jn = 0; jn < 2; ++jn) {
    const int ng   = n0 + waveN * 32 + jn * 16 + l16;
    const float bv = qkv_b[ng];
    const int sec  = (ng >= 2 * kC) ? 2 : ((ng >= kC) ? 1 : 0);
    const int cch  = ng - sec * kC;
    const int hh   = cch >> 6;
    const int dd   = cch & 63;
    _Float16* dst = (sec == 0) ? qbuf : ((sec == 1) ? kbuf : vbuf);
#pragma unroll
    for (int i = 0; i < 2; ++i) {
#pragma unroll
      for (int r = 0; r < 8; ++r) {
        const int mg = m0 + waveM * 32 + i * 16 + r + 8 * half;
        const int bb = mg >> 11;
        const int tt = mg & (kT - 1);
        dst[(((size_t)bb * kH + hh) * kT + tt) * kD + dd] =
            (_Float16)(acc[i][jn][r] + bv);
      }
    }
  }
}

// ---------------------------------------------------------------------------
// Kernel 2: flash attention. One wave per 16-query tile per (b,h).
// Block: 128 thr (4 waves). Streams keys in blocks of 32 (causal).
// ---------------------------------------------------------------------------
__global__ __launch_bounds__(128) void attn_kernel(
    const _Float16* __restrict__ qbuf, const _Float16* __restrict__ kbuf,
    const _Float16* __restrict__ vbuf, float* __restrict__ ybuf) {
  __shared__ __align__(16) _Float16 Kls[4][32][72];
  __shared__ __align__(16) _Float16 Vls[4][32][72];
  __shared__ __align__(16) _Float16 Pls[4][16][40];

  const int tid  = threadIdx.x;
  const int wv   = tid >> 5;
  const int lane = tid & 31;
  const int half = lane >> 4;
  const int l16  = lane & 15;

  const int wg  = blockIdx.x * 4 + wv;          // 0 .. B*H*(T/16)-1
  const int qt  = wg & 127;                     // T/16 = 128 query tiles
  const int rest = wg >> 7;
  const int hh  = rest % kH;
  const int bb  = rest / kH;
  const int q0  = qt * 16;

  const size_t base = ((size_t)bb * kH + hh) * (size_t)kT * kD;

  // Q fragments: rows q0..q0+15, D=64 -> two 16x32 A-frags.
  v16h qa[2];
#pragma unroll
  for (int s = 0; s < 2; ++s) {
    const size_t rowoff = base + (size_t)(q0 + l16) * kD + 32 * s;
#pragma unroll
    for (int j = 0; j < 8; ++j) {
      const int k = a_kidx(j, half);
      qa[s][2 * j]     = qbuf[rowoff + k];
      qa[s][2 * j + 1] = qbuf[rowoff + k + 1];
    }
  }

  float mrun[8], lrun[8];
#pragma unroll
  for (int r = 0; r < 8; ++r) { mrun[r] = -3.0e38f; lrun[r] = 0.0f; }
  v8f oacc[4] = {};

  const int ktmax = (q0 + 15) >> 5;  // inclusive; 32*kt <= q0 always holds
  for (int kt = 0; kt <= ktmax; ++kt) {
    const int k0 = kt * 32;

    // Stage K/V rows (key = k0+lane) into per-wave LDS via the async copy
    // engine: 8x b128 per lane per matrix, no VGPR round-trip.
    {
      const char* gk = (const char*)&kbuf[base + (size_t)(k0 + lane) * kD];
      const char* gv = (const char*)&vbuf[base + (size_t)(k0 + lane) * kD];
      const uint32_t lk = (uint32_t)(uintptr_t)&Kls[wv][lane][0];
      const uint32_t lv = (uint32_t)(uintptr_t)&Vls[wv][lane][0];
#pragma unroll
      for (int i = 0; i < 8; ++i) async_copy_b128(lk + 16 * i, gk + 16 * i);
#pragma unroll
      for (int i = 0; i < 8; ++i) async_copy_b128(lv + 16 * i, gv + 16 * i);
    }
    if (kt < ktmax) {  // speculative prefetch of next key block into caches
      __builtin_prefetch(&kbuf[base + (size_t)(k0 + 32 + lane) * kD], 0, 1);
      __builtin_prefetch(&vbuf[base + (size_t)(k0 + 32 + lane) * kD], 0, 1);
    }
    asm volatile("s_wait_asynccnt 0" ::: "memory");

    // S = Q @ K^T for 32 keys: two 16x16 f32 accumulators.
    v8f sacc[2] = {};
#pragma unroll
    for (int nt = 0; nt < 2; ++nt) {
#pragma unroll
      for (int s = 0; s < 2; ++s) {
        v16h kb;
        const int krow = nt * 16 + l16;
#pragma unroll
        for (int j = 0; j < 8; ++j) {
          const int d = b_kidx(j, half) + 32 * s;
          kb[2 * j]     = Kls[wv][krow][d];
          kb[2 * j + 1] = Kls[wv][krow][d + 1];
        }
        sacc[nt] = wmma_f16(qa[s], kb, sacc[nt]);
      }
    }

    // Online softmax update (fp32). Row m = r + 8*half lives across the
    // 16 lanes of this half; reduce with xor-shuffles (masks < 16 stay in half).
#pragma unroll
    for (int r = 0; r < 8; ++r) {
      const int q    = q0 + r + 8 * half;
      const int key0 = k0 + l16;
      const int key1 = k0 + 16 + l16;
      float s0 = sacc[0][r] * 0.125f;  // 1/sqrt(64)
      float s1 = sacc[1][r] * 0.125f;
      if (key0 > q) s0 = -3.0e38f;
      if (key1 > q) s1 = -3.0e38f;
      float vmx = fmaxf(s0, s1);
#pragma unroll
      for (int msk = 1; msk < 16; msk <<= 1)
        vmx = fmaxf(vmx, __shfl_xor(vmx, msk, 32));
      const float mnew  = fmaxf(mrun[r], vmx);
      const float alpha = __expf(mrun[r] - mnew);
      const float p0 = __expf(s0 - mnew);
      const float p1 = __expf(s1 - mnew);
      float rs = p0 + p1;
#pragma unroll
      for (int msk = 1; msk < 16; msk <<= 1) rs += __shfl_xor(rs, msk, 32);
      lrun[r] = lrun[r] * alpha + rs;
      mrun[r] = mnew;
#pragma unroll
      for (int nt = 0; nt < 4; ++nt) oacc[nt][r] *= alpha;
      Pls[wv][r + 8 * half][l16]      = (_Float16)p0;
      Pls[wv][r + 8 * half][16 + l16] = (_Float16)p1;
    }
    asm volatile("s_wait_dscnt 0" ::: "memory");

    // Re-layout P (16x32) as A-fragment, then O += P @ V.
    v16h pa;
#pragma unroll
    for (int j = 0; j < 8; ++j) {
      const int k = a_kidx(j, half);
      pa[2 * j]     = Pls[wv][l16][k];
      pa[2 * j + 1] = Pls[wv][l16][k + 1];
    }
#pragma unroll
    for (int nt = 0; nt < 4; ++nt) {
      v16h vb;
      const int dcol = nt * 16 + l16;
#pragma unroll
      for (int j = 0; j < 8; ++j) {
        const int kk = b_kidx(j, half);
        vb[2 * j]     = Vls[wv][kk][dcol];
        vb[2 * j + 1] = Vls[wv][kk + 1][dcol];
      }
      oacc[nt] = wmma_f16(pa, vb, oacc[nt]);
    }
    // Drain this iteration's DS reads before next iteration's async writes
    // can overwrite Kls/Vls (ASYNCcnt and DScnt are mutually unordered).
    asm volatile("s_wait_dscnt 0" ::: "memory");
  }

  // Normalize and write back to [B,T,C] (head-merged) fp32.
#pragma unroll
  for (int nt = 0; nt < 4; ++nt) {
#pragma unroll
    for (int r = 0; r < 8; ++r) {
      const int t  = q0 + r + 8 * half;
      const int cc = hh * kD + nt * 16 + l16;
      ybuf[((size_t)bb * kT + t) * kC + cc] = oacc[nt][r] / lrun[r];
    }
  }
}

// ---------------------------------------------------------------------------
// Kernel 3: out = y @ out_w^T + out_b   (M=4096, N=K=768), fp32 output.
// ---------------------------------------------------------------------------
__global__ __launch_bounds__(256) void out_proj_kernel(
    const float* __restrict__ y, const float* __restrict__ out_w,
    const float* __restrict__ out_b, float* __restrict__ out) {
  __shared__ __align__(16) _Float16 Xs[128][40];
  __shared__ __align__(16) _Float16 Ws[64][40];

  const int tid  = threadIdx.x;
  const int wave = tid >> 5;
  const int lane = tid & 31;
  const int half = lane >> 4;
  const int l16  = lane & 15;
  const int m0   = blockIdx.x * 128;
  const int n0   = blockIdx.y * 64;
  const int waveM = wave >> 1;
  const int waveN = wave & 1;

  v8f acc[2][2] = {};
  const int ldr  = tid >> 3;
  const int ldc4 = (tid & 7) * 4;

  for (int kk0 = 0; kk0 < kC; kk0 += 32) {
#pragma unroll
    for (int p = 0; p < 4; ++p) {
      const int row = p * 32 + ldr;
      const float4 v = *reinterpret_cast<const float4*>(
          &y[(size_t)(m0 + row) * kC + kk0 + ldc4]);
      Xs[row][ldc4 + 0] = (_Float16)v.x;
      Xs[row][ldc4 + 1] = (_Float16)v.y;
      Xs[row][ldc4 + 2] = (_Float16)v.z;
      Xs[row][ldc4 + 3] = (_Float16)v.w;
    }
#pragma unroll
    for (int p = 0; p < 2; ++p) {
      const int row = p * 32 + ldr;
      const float4 v = *reinterpret_cast<const float4*>(
          &out_w[(size_t)(n0 + row) * kC + kk0 + ldc4]);
      Ws[row][ldc4 + 0] = (_Float16)v.x;
      Ws[row][ldc4 + 1] = (_Float16)v.y;
      Ws[row][ldc4 + 2] = (_Float16)v.z;
      Ws[row][ldc4 + 3] = (_Float16)v.w;
    }
    __syncthreads();

    v16h afr[2], bfr[2];
#pragma unroll
    for (int i = 0; i < 2; ++i) {
      const int row = waveM * 32 + i * 16 + l16;
#pragma unroll
      for (int j = 0; j < 8; ++j) {
        const int k = a_kidx(j, half);
        afr[i][2 * j]     = Xs[row][k];
        afr[i][2 * j + 1] = Xs[row][k + 1];
      }
    }
#pragma unroll
    for (int i = 0; i < 2; ++i) {
      const int col = waveN * 32 + i * 16 + l16;
#pragma unroll
      for (int j = 0; j < 8; ++j) {
        const int k = b_kidx(j, half);
        bfr[i][2 * j]     = Ws[col][k];
        bfr[i][2 * j + 1] = Ws[col][k + 1];
      }
    }
#pragma unroll
    for (int i = 0; i < 2; ++i)
#pragma unroll
      for (int jn = 0; jn < 2; ++jn)
        acc[i][jn] = wmma_f16(afr[i], bfr[jn], acc[i][jn]);
    __syncthreads();
  }

#pragma unroll
  for (int jn = 0; jn < 2; ++jn) {
    const int ng   = n0 + waveN * 32 + jn * 16 + l16;
    const float bv = out_b[ng];
#pragma unroll
    for (int i = 0; i < 2; ++i) {
#pragma unroll
      for (int r = 0; r < 8; ++r) {
        const int mg = m0 + waveM * 32 + i * 16 + r + 8 * half;
        out[(size_t)mg * kC + ng] = acc[i][jn][r] + bv;
      }
    }
  }
}

// ---------------------------------------------------------------------------
extern "C" void kernel_launch(void* const* d_in, const int* in_sizes, int n_in,
                              void* d_out, int out_size, void* d_ws,
                              size_t ws_size, hipStream_t stream) {
  (void)in_sizes; (void)n_in; (void)out_size; (void)ws_size;
  const float* x     = (const float*)d_in[0];
  const float* qkv_w = (const float*)d_in[1];
  const float* qkv_b = (const float*)d_in[2];
  const float* out_w = (const float*)d_in[3];
  const float* out_b = (const float*)d_in[4];
  float* out = (float*)d_out;

  // Workspace layout (bytes): q/k/v f16 [B,H,T,D] then y f32 [B,T,C].
  const size_t szQKV = (size_t)kBatch * kH * kT * kD * sizeof(_Float16); // 6 MB
  char* ws = (char*)d_ws;
  _Float16* qbuf = (_Float16*)(ws);
  _Float16* kbuf = (_Float16*)(ws + szQKV);
  _Float16* vbuf = (_Float16*)(ws + 2 * szQKV);
  float*    ybuf = (float*)(ws + 3 * szQKV);  // 12 MB; total ~30 MB

  qkv_gemm_kernel<<<dim3(kM / 128, kNqkv / 64), 256, 0, stream>>>(
      x, qkv_w, qkv_b, qbuf, kbuf, vbuf);

  const int totalWaves = kBatch * kH * (kT / 16);  // 3072
  attn_kernel<<<dim3(totalWaves / 4), 128, 0, stream>>>(qbuf, kbuf, vbuf, ybuf);

  out_proj_kernel<<<dim3(kM / 128, kC / 64), 256, 0, stream>>>(
      ybuf, out_w, out_b, out);
}
